// MambaBlock_17961553231978
// MI455X (gfx1250) — matrix operand order repaired and checked by reading, not compiled, not run on hardware
//
#include <hip/hip_runtime.h>
#include <hip/hip_bf16.h>
#include <math.h>

// ---------------------------------------------------------------------------
// Mamba block for MI455X (gfx1250).
// bf16 WMMA GEMMs, fragment-major pre-packed operands, 2x2 register blocking
// per wave (32x32 C tile, 4 WMMA per 4 fragment loads) + fp32 VALU scan.
// ---------------------------------------------------------------------------

typedef __bf16 v16bf __attribute__((ext_vector_type(16)));
typedef float  v8f   __attribute__((ext_vector_type(8)));

#define DM     1024   // d_model
#define DI     2048   // d_inner
#define DSTATE 16
#define DCONV  4
#define DTRANK 64
#define BB     2
#define LL     1024
#define NROW   (BB * LL)   // 2048

__device__ __forceinline__ float silu_f(float v) {
    return v / (1.0f + __expf(-v));
}

__device__ __forceinline__ v8f wmma_bf16(v16bf a, v16bf b, v8f c) {
    return __builtin_amdgcn_wmma_f32_16x16x32_bf16(
        /*neg_a=*/false, a, /*neg_b=*/false, b,
        /*c_mod=*/(short)0, c, /*reuse_a=*/false, /*reuse_b=*/false);
}

// --------------------------- operand packing --------------------------------
// A-fragment (16x32, 16-bit): lane l: m=l&15, half=l>>4;
//   VGPR 0..3 hold K = half*8 + {0..7}; VGPR 4..7 hold K = 16+half*8+{0..7}.
// Packed [mt][kt][lane][16 bf16] -> per-lane fragment is one 32B load.
__global__ void pack_a_kernel(const float* __restrict__ src,
                              __bf16* __restrict__ dst,
                              int M, int K, int ld /* row stride, floats */) {
    int t = blockIdx.x * blockDim.x + threadIdx.x;     // (mt*KT + kt)*32 + lane
    int KT = K >> 5;
    int total = (M >> 4) * KT * 32;
    if (t >= total) return;
    int lane = t & 31;
    int kt   = (t >> 5) % KT;
    int mt   = (t >> 5) / KT;
    int m    = mt * 16 + (lane & 15);
    int half = lane >> 4;
    // two contiguous 8-float runs: [half*8, +8) and [16+half*8, +8)
    const float4* s = (const float4*)(src + (size_t)m * ld + kt * 32 + half * 8);
    float4 r0 = s[0], r1 = s[1];     // run 0
    float4 r2 = s[4], r3 = s[5];     // run 1 (16 floats further)
    v16bf f;
    f[0]  = (__bf16)r0.x; f[1]  = (__bf16)r0.y; f[2]  = (__bf16)r0.z; f[3]  = (__bf16)r0.w;
    f[4]  = (__bf16)r1.x; f[5]  = (__bf16)r1.y; f[6]  = (__bf16)r1.z; f[7]  = (__bf16)r1.w;
    f[8]  = (__bf16)r2.x; f[9]  = (__bf16)r2.y; f[10] = (__bf16)r2.z; f[11] = (__bf16)r2.w;
    f[12] = (__bf16)r3.x; f[13] = (__bf16)r3.y; f[14] = (__bf16)r3.z; f[15] = (__bf16)r3.w;
    *(v16bf*)(dst + (size_t)t * 16) = f;
}

// B-fragment (32x16): lane l: n=l&15, half=l>>4;
//   VGPR 0..3 hold K = half*8 + {0..7}; VGPR 4..7 hold K = 16+half*8+{0..7}, column n.
// Packed [nt][kt][lane][16 bf16].
__global__ void pack_b_kernel(const float* __restrict__ src,  // K x N row-major
                              __bf16* __restrict__ dst,
                              int K, int N) {
    int t = blockIdx.x * blockDim.x + threadIdx.x;     // (nt*KT + kt)*32 + lane
    int KT = K >> 5;
    int total = (N >> 4) * KT * 32;
    if (t >= total) return;
    int lane = t & 31;
    int kt   = (t >> 5) % KT;
    int nt   = (t >> 5) / KT;
    int n    = nt * 16 + (lane & 15);
    int half = lane >> 4;
    const float* s = src + (size_t)(kt * 32 + half * 8) * N + n;
    v16bf f;
    #pragma unroll
    for (int j = 0; j < 16; ++j)
        f[j] = (__bf16)s[(size_t)(j + (j & 8)) * N];
    *(v16bf*)(dst + (size_t)t * 16) = f;
}

// ------------------------------- WMMA GEMM ----------------------------------
// One wave computes a 32x32 C supertile (2x2 WMMA tiles): per K-step,
// 4 fragment loads (8x global_load_b128) feed 4 v_wmma issues.
__global__ __launch_bounds__(256)
void wmma_gemm_packed_kernel(const __bf16* __restrict__ Apack,
                             const __bf16* __restrict__ Bpack,
                             float* __restrict__ C,
                             int M, int N, int K) {
    const int lane = threadIdx.x & 31;
    const int wid  = blockIdx.x * (blockDim.x >> 5) + (threadIdx.x >> 5);
    const int sn   = N >> 5;                       // supertiles along N
    const int smt  = wid / sn;
    const int snt  = wid % sn;
    if (smt >= (M >> 5)) return;
    const int KT  = K >> 5;
    const int mt0 = smt * 2, nt0 = snt * 2;

    const v16bf* Ap0 = (const v16bf*)Apack + ((size_t)(mt0)     * KT * 32 + lane);
    const v16bf* Ap1 = (const v16bf*)Apack + ((size_t)(mt0 + 1) * KT * 32 + lane);
    const v16bf* Bp0 = (const v16bf*)Bpack + ((size_t)(nt0)     * KT * 32 + lane);
    const v16bf* Bp1 = (const v16bf*)Bpack + ((size_t)(nt0 + 1) * KT * 32 + lane);

    v8f acc00 = {0.f,0.f,0.f,0.f,0.f,0.f,0.f,0.f};
    v8f acc01 = acc00, acc10 = acc00, acc11 = acc00;

    #pragma unroll 2
    for (int kt = 0; kt < KT; ++kt) {
        // speculative prefetch of the A streams 4 K-steps ahead
        __builtin_prefetch((const void*)(Ap0 + (size_t)(kt + 4) * 32), 0, 0);
        __builtin_prefetch((const void*)(Ap1 + (size_t)(kt + 4) * 32), 0, 0);
        v16bf a0 = Ap0[(size_t)kt * 32];
        v16bf a1 = Ap1[(size_t)kt * 32];
        v16bf b0 = Bp0[(size_t)kt * 32];
        v16bf b1 = Bp1[(size_t)kt * 32];
        acc00 = wmma_bf16(a0, b0, acc00);
        acc01 = wmma_bf16(a0, b1, acc01);
        acc10 = wmma_bf16(a1, b0, acc10);
        acc11 = wmma_bf16(a1, b1, acc11);
    }

    // C layout per tile: VGPR v -> row half*8+v, lane -> column lane&15.
    const int col  = lane & 15;
    const int half = lane >> 4;
    const int r0   = mt0 * 16 + half * 8;
    const int c0   = nt0 * 16 + col;
    #pragma unroll
    for (int v = 0; v < 8; ++v) {
        C[(size_t)(r0 + v)      * N + c0]      = acc00[v];
        C[(size_t)(r0 + v)      * N + c0 + 16] = acc01[v];
        C[(size_t)(r0 + 16 + v) * N + c0]      = acc10[v];
        C[(size_t)(r0 + 16 + v) * N + c0 + 16] = acc11[v];
    }
}

// --------------------------- depthwise conv + SiLU ---------------------------
__global__ void conv_silu_kernel(const float* __restrict__ xz,
                                 const float* __restrict__ conv_w,
                                 const float* __restrict__ conv_b,
                                 float* __restrict__ xc_f32) {
    int idx = blockIdx.x * blockDim.x + threadIdx.x;   // over BB*LL*DI
    if (idx >= BB * LL * DI) return;
    int d = idx % DI;
    int l = (idx / DI) % LL;
    int b = idx / (DI * LL);

    float acc = conv_b[d];
    #pragma unroll
    for (int j = 0; j < DCONV; ++j) {
        int t = l - (DCONV - 1) + j;
        if (t >= 0)
            acc += conv_w[d * DCONV + j] *
                   xz[((size_t)b * LL + t) * (2 * DI) + d];
    }
    xc_f32[idx] = silu_f(acc);
}

// ------------------------------ selective scan -------------------------------
// One lane per (b, d) channel: h[16] in VGPRs, 1024 sequential steps.
// B/C loads are wave-uniform; dt/xc/z loads coalesced across lanes.
__global__ __launch_bounds__(256)
void scan_kernel(const float* __restrict__ dtraw,   // [NROW][DI]
                 const float* __restrict__ b_dt,    // [DI]
                 const float* __restrict__ xc,      // [NROW][DI]
                 const float* __restrict__ dbl,     // [NROW][96]
                 const float* __restrict__ A_log,   // [DI][16]
                 const float* __restrict__ D_param, // [DI]
                 const float* __restrict__ xz,      // [NROW][2*DI]
                 float* __restrict__ ypre) {        // [NROW][DI]
    const int d = blockIdx.x * blockDim.x + threadIdx.x;   // 0..DI-1
    const int b = blockIdx.y;

    float Aneg[DSTATE];
    #pragma unroll
    for (int n = 0; n < DSTATE; ++n)
        Aneg[n] = -__expf(A_log[d * DSTATE + n]);

    float h[DSTATE];
    #pragma unroll
    for (int n = 0; n < DSTATE; ++n) h[n] = 0.f;

    const float bdt = b_dt[d];
    const float Dp  = D_param[d];

    for (int l = 0; l < LL; ++l) {
        const size_t row = (size_t)b * LL + l;
        // prefetch the sequential streams 8 timesteps ahead
        __builtin_prefetch((const void*)(dtraw + (row + 8) * DI + d), 0, 0);
        __builtin_prefetch((const void*)(xc    + (row + 8) * DI + d), 0, 0);
        __builtin_prefetch((const void*)(xz + (row + 8) * (2 * DI) + DI + d), 0, 0);

        float dtv = dtraw[row * DI + d] + bdt;
        dtv = (dtv > 20.f) ? dtv : log1pf(__expf(dtv));    // softplus
        const float xcv = xc[row * DI + d];
        const float dtx = dtv * xcv;
        const float* dblrow = dbl + row * 96;

        float y = 0.f;
        #pragma unroll
        for (int n = 0; n < DSTATE; ++n) {
            float dA = __expf(dtv * Aneg[n]);
            float Bn = dblrow[DTRANK + n];
            float Cn = dblrow[DTRANK + DSTATE + n];
            h[n] = dA * h[n] + dtx * Bn;
            y += h[n] * Cn;
        }
        float zv = xz[row * (2 * DI) + DI + d];
        ypre[row * DI + d] = (y + xcv * Dp) * silu_f(zv);
    }
}

// ------------------------------- launch --------------------------------------
extern "C" void kernel_launch(void* const* d_in, const int* in_sizes, int n_in,
                              void* d_out, int out_size, void* d_ws, size_t ws_size,
                              hipStream_t stream) {
    const float* x      = (const float*)d_in[0];
    const float* W_in   = (const float*)d_in[1];
    const float* conv_w = (const float*)d_in[2];
    const float* conv_b = (const float*)d_in[3];
    const float* W_x    = (const float*)d_in[4];
    const float* W_dt   = (const float*)d_in[5];
    const float* b_dt   = (const float*)d_in[6];
    const float* A_log  = (const float*)d_in[7];
    const float* D_par  = (const float*)d_in[8];
    const float* W_out  = (const float*)d_in[9];
    float* out = (float*)d_out;

    char* ws = (char*)d_ws;
    size_t off = 0;
    auto alloc = [&](size_t bytes) -> void* {
        void* p = ws + off;
        off = (off + bytes + 255) & ~(size_t)255;
        return p;
    };

    // f32 intermediates
    float* xz    = (float*)alloc((size_t)NROW * 2 * DI * 4);  // 2048 x 4096
    float* xc    = (float*)alloc((size_t)NROW * DI * 4);      // 2048 x 2048
    float* dbl   = (float*)alloc((size_t)NROW * 96 * 4);      // 2048 x 96
    float* dtraw = (float*)alloc((size_t)NROW * DI * 4);      // 2048 x 2048
    float* ypre  = (float*)alloc((size_t)NROW * DI * 4);      // 2048 x 2048
    // fragment-packed bf16 operands
    __bf16* xP    = (__bf16*)alloc((size_t)NROW * DM * 2);
    __bf16* WinP  = (__bf16*)alloc((size_t)DM * 2 * DI * 2);
    __bf16* xcP   = (__bf16*)alloc((size_t)NROW * DI * 2);
    __bf16* WxP   = (__bf16*)alloc((size_t)DI * 96 * 2);
    __bf16* dtinP = (__bf16*)alloc((size_t)NROW * DTRANK * 2);
    __bf16* WdtP  = (__bf16*)alloc((size_t)DTRANK * DI * 2);
    __bf16* ypreP = (__bf16*)alloc((size_t)NROW * DI * 2);
    __bf16* WoutP = (__bf16*)alloc((size_t)DI * DM * 2);

    const int CT = 256;
    auto cdiv = [](int a, int b) { return (a + b - 1) / b; };
    auto gemm_grid = [&](int M, int N) { return cdiv((M >> 5) * (N >> 5), 8); };

    // ---- pack weights ----
    pack_b_kernel<<<cdiv((2*DI/16)*(DM/32)*32, CT), CT, 0, stream>>>(W_in,  WinP,  DM,     2*DI);
    pack_b_kernel<<<cdiv((96/16)*(DI/32)*32,   CT), CT, 0, stream>>>(W_x,   WxP,   DI,     96);
    pack_b_kernel<<<cdiv((DI/16)*(DTRANK/32)*32, CT), CT, 0, stream>>>(W_dt, WdtP,  DTRANK, DI);
    pack_b_kernel<<<cdiv((DM/16)*(DI/32)*32,   CT), CT, 0, stream>>>(W_out, WoutP, DI,     DM);

    // ---- GEMM1: xz = x @ W_in  (2048 x 1024 x 4096) ----
    pack_a_kernel<<<cdiv((NROW/16)*(DM/32)*32, CT), CT, 0, stream>>>(x, xP, NROW, DM, DM);
    wmma_gemm_packed_kernel<<<gemm_grid(NROW, 2*DI), 256, 0, stream>>>(
        xP, WinP, xz, NROW, 2*DI, DM);

    // ---- depthwise conv + SiLU -> xc ----
    conv_silu_kernel<<<cdiv(BB * LL * DI, CT), CT, 0, stream>>>(xz, conv_w, conv_b, xc);

    // ---- GEMM2: dbl = xc @ W_x  (2048 x 2048 x 96) ----
    pack_a_kernel<<<cdiv((NROW/16)*(DI/32)*32, CT), CT, 0, stream>>>(xc, xcP, NROW, DI, DI);
    wmma_gemm_packed_kernel<<<gemm_grid(NROW, 96), 256, 0, stream>>>(
        xcP, WxP, dbl, NROW, 96, DI);

    // ---- GEMM3: dtraw = dbl[:, :64] @ W_dt  (2048 x 64 x 2048) ----
    pack_a_kernel<<<cdiv((NROW/16)*(DTRANK/32)*32, CT), CT, 0, stream>>>(dbl, dtinP, NROW, DTRANK, 96);
    wmma_gemm_packed_kernel<<<gemm_grid(NROW, DI), 256, 0, stream>>>(
        dtinP, WdtP, dtraw, NROW, DI, DTRANK);

    // ---- selective scan + gating -> ypre ----
    {
        dim3 grid(DI / 256, BB);
        scan_kernel<<<grid, 256, 0, stream>>>(dtraw, b_dt, xc, dbl,
                                              A_log, D_par, xz, ypre);
    }

    // ---- GEMM4: out = ypre @ W_out  (2048 x 2048 x 1024) ----
    pack_a_kernel<<<cdiv((NROW/16)*(DI/32)*32, CT), CT, 0, stream>>>(ypre, ypreP, NROW, DI, DI);
    wmma_gemm_packed_kernel<<<gemm_grid(NROW, DM), 256, 0, stream>>>(
        ypreP, WoutP, out, NROW, DM, DI);
}
